// DecoderModel_13194139533905
// MI455X (gfx1250) — matrix operand order repaired and checked by reading, not compiled, not run on hardware
//
#include <hip/hip_runtime.h>
#include <hip/hip_bf16.h>
#include <math.h>

// ---------------------------------------------------------------------------
// Model dims (fixed by the reference)
// ---------------------------------------------------------------------------
namespace {
constexpr int Bc  = 2;
constexpr int Sc  = 512;
constexpr int Dc  = 1024;
constexpr int Hc  = 16;
constexpr int Lc  = 12;
constexpr int Vc  = 50257;
constexpr int HDc = 64;          // head dim
constexpr int DFFc = 4096;
constexpr int D3  = 3 * Dc;      // qkv width
constexpr int KVc = Sc + 1;      // kv length incl. image token
}

typedef __attribute__((ext_vector_type(16))) __bf16 v16bf;
typedef __attribute__((ext_vector_type(8)))  __bf16 v8bf;
typedef __attribute__((ext_vector_type(8)))  float  v8f;

// ---------------------------------------------------------------------------
// Helpers
// ---------------------------------------------------------------------------
__device__ __forceinline__ float wave_sum32(float v) {
#pragma unroll
  for (int m = 16; m > 0; m >>= 1) v += __shfl_xor(v, m, 32);
  return v;
}
__device__ __forceinline__ float wave_max32(float v) {
#pragma unroll
  for (int m = 16; m > 0; m >>= 1) v = fmaxf(v, __shfl_xor(v, m, 32));
  return v;
}
__device__ __forceinline__ float gelu_tanh(float x) {
  const float c = 0.7978845608028654f;  // sqrt(2/pi)
  return 0.5f * x * (1.0f + tanhf(c * (x + 0.044715f * x * x * x)));
}
// pack two fp32 into adjacent bf16 halves of one dword (v_cvt_pk_bf16_f32)
__device__ __forceinline__ unsigned pack_bf16x2(float a, float b) {
  union { __bf16 h[2]; unsigned u; } p;
  p.h[0] = (__bf16)a; p.h[1] = (__bf16)b;
  return p.u;
}

// ---------------------------------------------------------------------------
// Embedding: h[b,s,:] = wte[ids[b,s],:] + wte[s,:]   (source uses wte for pos)
// ---------------------------------------------------------------------------
__global__ __launch_bounds__(256) void embed_k(const int* __restrict__ ids,
                                               const float* __restrict__ wte,
                                               float* __restrict__ h) {
  const int bs = blockIdx.x;               // 0 .. B*S-1
  const int s  = bs % Sc;
  const int tok = ids[bs];
  const float* te = wte + (size_t)tok * Dc;
  const float* pe = wte + (size_t)s * Dc;
  float* out = h + (size_t)bs * Dc;
  for (int d = threadIdx.x; d < Dc; d += 256) out[d] = te[d] + pe[d];
}

// ---------------------------------------------------------------------------
// LayerNorm over last dim (D=1024), jnp.var semantics (ddof=0), eps 1e-5
// ---------------------------------------------------------------------------
__global__ __launch_bounds__(256) void layernorm_k(const float* __restrict__ x,
                                                   const float* __restrict__ g,
                                                   const float* __restrict__ b,
                                                   float* __restrict__ y) {
  const int row = blockIdx.x;
  const float* xr = x + (size_t)row * Dc;
  float s = 0.f, s2 = 0.f;
  for (int d = threadIdx.x; d < Dc; d += 256) {
    float v = xr[d];
    s += v; s2 += v * v;
  }
  __shared__ float red[2][8];
  const int wave = threadIdx.x >> 5, lane = threadIdx.x & 31;
  s  = wave_sum32(s);
  s2 = wave_sum32(s2);
  if (lane == 0) { red[0][wave] = s; red[1][wave] = s2; }
  __syncthreads();
  float S = 0.f, S2 = 0.f;
#pragma unroll
  for (int i = 0; i < 8; ++i) { S += red[0][i]; S2 += red[1][i]; }
  const float mean = S * (1.0f / Dc);
  const float var  = S2 * (1.0f / Dc) - mean * mean;
  const float inv  = rsqrtf(var + 1e-5f);
  float* yr = y + (size_t)row * Dc;
  for (int d = threadIdx.x; d < Dc; d += 256)
    yr[d] = (xr[d] - mean) * inv * g[d] + b[d];
}

// ---------------------------------------------------------------------------
// Small dense layer for the image path (R rows, fp32): out = act(in @ W + b)
// ACT: 0 = none, 2 = relu.  W is [K,N] row-major.
// ---------------------------------------------------------------------------
template <int ACT>
__global__ __launch_bounds__(64) void linear_small(const float* __restrict__ in,
                                                   const float* __restrict__ W,
                                                   const float* __restrict__ bias,
                                                   float* __restrict__ out,
                                                   int N, int K) {
  const int n = blockIdx.x * 64 + threadIdx.x;
  const int r = blockIdx.y;
  if (n >= N) return;
  const float* a = in + (size_t)r * K;
  float acc = bias ? bias[n] : 0.f;
  for (int k = 0; k < K; ++k) acc = fmaf(a[k], W[(size_t)k * N + n], acc);
  if (ACT == 2) acc = fmaxf(acc, 0.f);
  out[(size_t)r * N + n] = acc;
}

// ---------------------------------------------------------------------------
// WMMA bf16 GEMM:  C[M,N] = epilogue(A[M,K] @ B + bias [+ res])
//   TRANS_B=false : B is [K,N] row-major   (RAGGED must be false; N % 128 == 0)
//   TRANS_B=true  : B is [N,K] row-major   (tied lm head: wte, ragged N ok)
// Block: 256 threads = 8 waves, tile 128x128, K-step 32.
// Double-buffered LDS + register prefetch: one barrier per K-step, global
// loads for step kt+1 issued before the WMMAs of step kt.
// Non-trans B staging transposes [k][n] -> [n][k] in registers: adjacent-k row
// pairs are fetched so the fp32->bf16 conversion packs k-pairs into full
// dwords (v_cvt_pk_bf16_f32) and stores are aligned b32 (no b16 scatter).
// Wave w covers rows (w&3)*32..+31, cols (w>>2)*64..+63  => 2x4 accum tiles.
// Fragments: A/B in contiguous k order (lane half h holds k 16h..16h+15), f32 C 16x16 per CDNA5 ISA 7.12.2.
// ---------------------------------------------------------------------------
template <int ACT, bool HAS_BIAS, bool HAS_RES, bool TRANS_B, bool RAGGED>
__global__ __launch_bounds__(256) void gemm_wmma(const float* __restrict__ A,
                                                 const float* __restrict__ Bm,
                                                 const float* __restrict__ bias,
                                                 const float* __restrict__ res,
                                                 float* __restrict__ C,
                                                 int M, int N, int K) {
  __shared__ __align__(32) __bf16 As[2][128 * 32];  // [m][k], k contiguous
  __shared__ __align__(32) __bf16 Bs[2][128 * 32];  // [n][k], k contiguous

  const int t    = threadIdx.x;
  const int wave = t >> 5;
  const int lane = t & 31;
  const int m0 = blockIdx.y * 128;
  const int n0 = blockIdx.x * 128;
  const int wm = (wave & 3) * 32;
  const int wn = (wave >> 2) * 64;

  // staging coordinates (per thread, constant across K-steps)
  const int ar = t >> 3;         // 0..31 : row within a 32-row pass (A / trans-B)
  const int ak = (t & 7) * 4;    // k offset (A / trans-B)
  const int nq = (t & 31) * 4;   // non-trans B: 4-column group
  const int kp = (t >> 5) * 2;   // non-trans B: k-pair base (0,2,..,14)

  float4 pa[4], pb[4];

  // ---- issue global loads for the tile starting at k0 (no waits here)
  auto fetch = [&](int k0) {
#pragma unroll
    for (int p = 0; p < 4; ++p)
      pa[p] = *(const float4*)(A + (size_t)(m0 + p * 32 + ar) * K + (k0 + ak));
    if (!TRANS_B) {
      // adjacent-k row pairs at the same column group (coalesced across lanes)
      pb[0] = *(const float4*)(Bm + (size_t)(k0 + kp + 0)  * N + (n0 + nq));
      pb[1] = *(const float4*)(Bm + (size_t)(k0 + kp + 1)  * N + (n0 + nq));
      pb[2] = *(const float4*)(Bm + (size_t)(k0 + kp + 16) * N + (n0 + nq));
      pb[3] = *(const float4*)(Bm + (size_t)(k0 + kp + 17) * N + (n0 + nq));
    } else {
#pragma unroll
      for (int p = 0; p < 4; ++p) {
        const int gn = n0 + p * 32 + ar;
        if (!RAGGED || gn < N)
          pb[p] = *(const float4*)(Bm + (size_t)gn * K + (k0 + ak));
        else
          pb[p] = make_float4(0.f, 0.f, 0.f, 0.f);
      }
    }
  };

  // ---- convert fp32 -> bf16 and store into LDS buffer `buf`
  auto stage = [&](int buf) {
    __bf16* Ad = As[buf];
    __bf16* Bd = Bs[buf];
#pragma unroll
    for (int p = 0; p < 4; ++p) {
      __bf16* d = Ad + (p * 32 + ar) * 32 + ak;
      d[0] = (__bf16)pa[p].x; d[1] = (__bf16)pa[p].y;
      d[2] = (__bf16)pa[p].z; d[3] = (__bf16)pa[p].w;
    }
    if (!TRANS_B) {
      // [n][k] with k-pairs packed into dwords: 8 aligned b32 stores
      *(unsigned*)(Bd + (nq + 0) * 32 + kp)      = pack_bf16x2(pb[0].x, pb[1].x);
      *(unsigned*)(Bd + (nq + 1) * 32 + kp)      = pack_bf16x2(pb[0].y, pb[1].y);
      *(unsigned*)(Bd + (nq + 2) * 32 + kp)      = pack_bf16x2(pb[0].z, pb[1].z);
      *(unsigned*)(Bd + (nq + 3) * 32 + kp)      = pack_bf16x2(pb[0].w, pb[1].w);
      *(unsigned*)(Bd + (nq + 0) * 32 + kp + 16) = pack_bf16x2(pb[2].x, pb[3].x);
      *(unsigned*)(Bd + (nq + 1) * 32 + kp + 16) = pack_bf16x2(pb[2].y, pb[3].y);
      *(unsigned*)(Bd + (nq + 2) * 32 + kp + 16) = pack_bf16x2(pb[2].z, pb[3].z);
      *(unsigned*)(Bd + (nq + 3) * 32 + kp + 16) = pack_bf16x2(pb[2].w, pb[3].w);
    } else {
#pragma unroll
      for (int p = 0; p < 4; ++p) {
        __bf16* d = Bd + (p * 32 + ar) * 32 + ak;
        d[0] = (__bf16)pb[p].x; d[1] = (__bf16)pb[p].y;
        d[2] = (__bf16)pb[p].z; d[3] = (__bf16)pb[p].w;
      }
    }
  };

  v8f acc[2][4] = {};

  const int ksteps = K >> 5;  // K is a multiple of 32 for all uses here

  fetch(0);
  stage(0);
  __syncthreads();

  // fragment coordinates (A/B: contiguous k, lane half h holds k 16h..16h+15)
  const int am    = lane & 15;               // A: M row
  const int khalf = (lane < 16) ? 0 : 8;     // A: K half select
  const int bn    = lane & 15;               // B: N col
  const int bko   = (lane < 16) ? 0 : 16;    // B: K half select

  for (int kt = 0; kt < ksteps; ++kt) {
    if (kt + 1 < ksteps) fetch((kt + 1) << 5);  // prefetch next tile (async)

    const __bf16* Ab = As[kt & 1];
    const __bf16* Bb = Bs[kt & 1];

    v16bf afrag[2];
#pragma unroll
    for (int r = 0; r < 2; ++r) {
      const __bf16* rowp = Ab + (wm + r * 16 + am) * 32;
      const v8bf lo = *(const v8bf*)(rowp + 2 * khalf);
      const v8bf hi = *(const v8bf*)(rowp + 2 * khalf + 8);
      v16bf a;
#pragma unroll
      for (int i = 0; i < 8; ++i) { a[i] = lo[i]; a[i + 8] = hi[i]; }
      afrag[r] = a;
    }
    v16bf bfrag[4];
#pragma unroll
    for (int c = 0; c < 4; ++c)
      bfrag[c] = *(const v16bf*)(Bb + (wn + c * 16 + bn) * 32 + bko);

#pragma unroll
    for (int r = 0; r < 2; ++r)
#pragma unroll
      for (int c = 0; c < 4; ++c)
        acc[r][c] = __builtin_amdgcn_wmma_f32_16x16x32_bf16(
            false, afrag[r], false, bfrag[c], (short)0, acc[r][c], false, false);

    if (kt + 1 < ksteps) stage((kt + 1) & 1);  // write the other buffer
    __syncthreads();
  }

  // ---- epilogue.  f32 C 16x16: VGPR r holds M=r (lanes 0-15) / M=r+8 (16-31), N=lane&15.
  const int cn  = lane & 15;
  const int mhi = (lane >> 4) << 3;
#pragma unroll
  for (int r = 0; r < 2; ++r) {
#pragma unroll
    for (int c = 0; c < 4; ++c) {
#pragma unroll
      for (int reg = 0; reg < 8; ++reg) {
        const int m = m0 + wm + r * 16 + reg + mhi;
        const int n = n0 + wn + c * 16 + cn;
        if (!RAGGED || n < N) {
          float v = acc[r][c][reg];
          if (HAS_BIAS) v += bias[n];
          if (ACT == 1) v = gelu_tanh(v);
          if (HAS_RES) v += res[(size_t)m * N + n];
          C[(size_t)m * N + n] = v;
        }
      }
    }
  }
}

// ---------------------------------------------------------------------------
// Attention: one wave per query, 8 queries per block, kv = S+1 (image token).
// scores = q.k/8 with causal(j<=i+1) + padding mask; softmax; out = p.v
// K/V tiles are staged through LDS (row pad 65 -> conflict-free).
// ---------------------------------------------------------------------------
__global__ __launch_bounds__(256) void attn_k(const float* __restrict__ qkv,
                                              const float* __restrict__ ki,
                                              const float* __restrict__ vi,
                                              const float* __restrict__ amask,
                                              float* __restrict__ outp) {
  __shared__ float qs[8][HDc];        // per-wave query
  __shared__ float sc[8][KVc + 7];    // scores/probs
  __shared__ float kv[32][65];        // staged K (then V) tile

  const int qt = blockIdx.x % (Sc / 8);
  const int h  = (blockIdx.x / (Sc / 8)) % Hc;
  const int b  = blockIdx.x / ((Sc / 8) * Hc);
  const int wave = threadIdx.x >> 5;
  const int lane = threadIdx.x & 31;
  const int i = qt * 8 + wave;  // query index

  // load queries
  const size_t qbase = ((size_t)(b * Sc + i)) * D3 + h * HDc;
  qs[wave][lane]      = qkv[qbase + lane];
  qs[wave][lane + 32] = qkv[qbase + lane + 32];
  __syncthreads();

  const int ntiles = (KVc + 31) / 32;  // 17

  // ---- pass 1: scores
  for (int jt = 0; jt < ntiles; ++jt) {
    const int j0 = jt * 32;
    {  // cooperative K tile load: 32 keys x 64 dims, 8 floats/thread
      const int jj = threadIdx.x >> 3;
      const int db = (threadIdx.x & 7) * 8;
      const int j = j0 + jj;
      if (j < KVc) {
        const float* kr = (j == 0)
            ? (ki + (size_t)b * Dc + h * HDc)
            : (qkv + ((size_t)(b * Sc + (j - 1))) * D3 + Dc + h * HDc);
#pragma unroll
        for (int u = 0; u < 8; ++u) kv[jj][db + u] = kr[db + u];
      }
    }
    __syncthreads();
    const int j = j0 + lane;
    if (j < KVc) {
      float s = 0.f;
#pragma unroll
      for (int d = 0; d < HDc; ++d) s = fmaf(qs[wave][d], kv[lane][d], s);
      float val = (j <= i + 1) ? s * 0.125f : -10000.0f;         // /sqrt(64), causal
      if (j > 0) val += (1.0f - amask[b * Sc + (j - 1)]) * -10000.0f;
      sc[wave][j] = val;
    }
    __syncthreads();
  }

  // ---- softmax (per wave over its own row)
  float mx = -3.4e38f;
  for (int j = lane; j < KVc; j += 32) mx = fmaxf(mx, sc[wave][j]);
  mx = wave_max32(mx);
  float sum = 0.f;
  for (int j = lane; j < KVc; j += 32) {
    float e = __expf(sc[wave][j] - mx);
    sc[wave][j] = e;
    sum += e;
  }
  sum = wave_sum32(sum);
  const float rinv = 1.0f / sum;
  for (int j = lane; j < KVc; j += 32) sc[wave][j] *= rinv;
  __syncthreads();

  // ---- pass 2: out = p . V  (2 dims per lane)
  float o0 = 0.f, o1 = 0.f;
  for (int jt = 0; jt < ntiles; ++jt) {
    const int j0 = jt * 32;
    {  // cooperative V tile load
      const int jj = threadIdx.x >> 3;
      const int db = (threadIdx.x & 7) * 8;
      const int j = j0 + jj;
      if (j < KVc) {
        const float* vr = (j == 0)
            ? (vi + (size_t)b * Dc + h * HDc)
            : (qkv + ((size_t)(b * Sc + (j - 1))) * D3 + 2 * Dc + h * HDc);
#pragma unroll
        for (int u = 0; u < 8; ++u) kv[jj][db + u] = vr[db + u];
      }
    }
    __syncthreads();
#pragma unroll
    for (int jj = 0; jj < 32; ++jj) {
      const int j = j0 + jj;
      if (j < KVc) {
        const float p = sc[wave][j];
        o0 = fmaf(p, kv[jj][lane], o0);
        o1 = fmaf(p, kv[jj][lane + 32], o1);
      }
    }
    __syncthreads();
  }
  const size_t obase = ((size_t)(b * Sc + i)) * Dc + h * HDc;
  outp[obase + lane]      = o0;
  outp[obase + lane + 32] = o1;
}

// ---------------------------------------------------------------------------
// Host orchestration
// ---------------------------------------------------------------------------
extern "C" void kernel_launch(void* const* d_in, const int* in_sizes, int n_in,
                              void* d_out, int out_size, void* d_ws, size_t ws_size,
                              hipStream_t stream) {
  (void)in_sizes; (void)n_in; (void)out_size; (void)ws_size;
  const int*   ids   = (const int*)  d_in[0];
  const float* amask = (const float*)d_in[1];
  const float* imgh  = (const float*)d_in[2];
  const float* wte   = (const float*)d_in[3];
  const float* ftW1  = (const float*)d_in[4];
  const float* ftb1  = (const float*)d_in[5];
  const float* ftW2  = (const float*)d_in[6];
  const float* ftb2  = (const float*)d_in[7];
  const float* ln1g  = (const float*)d_in[8];
  const float* ln1b  = (const float*)d_in[9];
  const float* Wattn = (const float*)d_in[10];
  const float* battn = (const float*)d_in[11];
  const float* Wuk   = (const float*)d_in[12];
  const float* buk   = (const float*)d_in[13];
  const float* Wuv   = (const float*)d_in[14];
  const float* buv   = (const float*)d_in[15];
  const float* Wproj = (const float*)d_in[16];
  const float* bproj = (const float*)d_in[17];
  const float* ln2g  = (const float*)d_in[18];
  const float* ln2b  = (const float*)d_in[19];
  const float* Wfc   = (const float*)d_in[20];
  const float* bfc   = (const float*)d_in[21];
  const float* Wfc2  = (const float*)d_in[22];
  const float* bfc2  = (const float*)d_in[23];
  const float* lnfg  = (const float*)d_in[24];
  const float* lnfb  = (const float*)d_in[25];
  float* out = (float*)d_out;

  // workspace layout (~43 MB of fp32)
  float* ws = (float*)d_ws;
  const size_t MS = (size_t)Bc * Sc;  // 1024 rows
  float* h     = ws;  ws += MS * Dc;
  float* x     = ws;  ws += MS * Dc;
  float* qkv   = ws;  ws += MS * D3;
  float* ffn   = ws;  ws += MS * DFFc;
  float* attno = ws;  ws += MS * Dc;
  float* imid  = ws;  ws += (size_t)Bc * Dc;
  float* img   = ws;  ws += (size_t)Bc * Dc;
  float* kiv   = ws;  ws += (size_t)Bc * Dc;
  float* viv   = ws;  ws += (size_t)Bc * Dc;

  // embeddings (token + wte-as-positional, faithful to source)
  embed_k<<<(int)MS, 256, 0, stream>>>(ids, wte, h);

  // image feature MLP: relu(img @ W1 + b1) @ W2 + b2
  {
    dim3 g((Dc + 63) / 64, Bc);
    linear_small<2><<<g, 64, 0, stream>>>(imgh, ftW1, ftb1, imid, Dc, Dc);
    linear_small<0><<<g, 64, 0, stream>>>(imid, ftW2, ftb2, img, Dc, Dc);
  }

  for (int l = 0; l < Lc; ++l) {
    layernorm_k<<<(int)MS, 256, 0, stream>>>(h, ln1g + (size_t)l * Dc, ln1b + (size_t)l * Dc, x);

    // qkv = x @ Wattn + battn
    gemm_wmma<0, true, false, false, false><<<dim3(D3 / 128, (int)MS / 128), 256, 0, stream>>>(
        x, Wattn + (size_t)l * Dc * D3, battn + (size_t)l * D3, nullptr, qkv, (int)MS, D3, Dc);

    // image-token k/v
    {
      dim3 g((Dc + 63) / 64, Bc);
      linear_small<0><<<g, 64, 0, stream>>>(img, Wuk + (size_t)l * Dc * Dc, buk + (size_t)l * Dc, kiv, Dc, Dc);
      linear_small<0><<<g, 64, 0, stream>>>(img, Wuv + (size_t)l * Dc * Dc, buv + (size_t)l * Dc, viv, Dc, Dc);
    }

    attn_k<<<Bc * Hc * (Sc / 8), 256, 0, stream>>>(qkv, kiv, viv, amask, attno);

    // h = h + attno @ Wproj + bproj
    gemm_wmma<0, true, true, false, false><<<dim3(Dc / 128, (int)MS / 128), 256, 0, stream>>>(
        attno, Wproj + (size_t)l * Dc * Dc, bproj + (size_t)l * Dc, h, h, (int)MS, Dc, Dc);

    layernorm_k<<<(int)MS, 256, 0, stream>>>(h, ln2g + (size_t)l * Dc, ln2b + (size_t)l * Dc, x);

    // ffn = gelu(x @ Wfc + bfc)
    gemm_wmma<1, true, false, false, false><<<dim3(DFFc / 128, (int)MS / 128), 256, 0, stream>>>(
        x, Wfc + (size_t)l * Dc * DFFc, bfc + (size_t)l * DFFc, nullptr, ffn, (int)MS, DFFc, Dc);

    // h = h + ffn @ Wfc2 + bfc2
    gemm_wmma<0, true, true, false, false><<<dim3(Dc / 128, (int)MS / 128), 256, 0, stream>>>(
        ffn, Wfc2 + (size_t)l * DFFc * Dc, bfc2 + (size_t)l * Dc, h, h, (int)MS, Dc, DFFc);
  }

  layernorm_k<<<(int)MS, 256, 0, stream>>>(h, lnfg, lnfb, x);

  // logits = x @ wte^T   (B transposed, ragged N=50257)
  gemm_wmma<0, false, false, true, true><<<dim3((Vc + 127) / 128, (int)MS / 128), 256, 0, stream>>>(
      x, wte, nullptr, nullptr, out, (int)MS, Vc, Dc);
}
